// SAGE_2869038153994
// MI455X (gfx1250) — matrix operand order repaired and checked by reading, not compiled, not run on hardware
//
#include <hip/hip_runtime.h>
#include <stdint.h>

// ---------------- problem constants ----------------
#define NN   100000
#define EE   1600000
#define DIN  128
#define DH   128
#define DOUT 64
#define KTOT 256          // concat of [mean(128) | self(128)]
#define APITCH 264        // padded LDS pitch (halves) to avoid bank conflicts

typedef __attribute__((ext_vector_type(16))) _Float16 v16h;
typedef __attribute__((ext_vector_type(8)))  float    v8f;

// ---------------- JAX threefry2x32 dropout mask (key(42)) ----------------
__device__ __forceinline__ unsigned rotl32(unsigned x, int r) {
  return (x << r) | (x >> (32 - r));
}

__device__ __forceinline__ bool dropout_keep(unsigned idx) {
  const unsigned half = (unsigned)(NN * DH / 2);  // 6,400,000
  unsigned x0, x1;
  bool lowHalf = idx < half;
  if (lowHalf) { x0 = idx;        x1 = idx + half; }
  else         { x0 = idx - half; x1 = idx;        }
  unsigned ks[3] = {0u, 42u, 0u ^ 42u ^ 0x1BD11BDAu};
  x0 += ks[0]; x1 += ks[1];
  const int rotA[4] = {13, 15, 26, 6};
  const int rotB[4] = {17, 29, 16, 24};
#pragma unroll
  for (int g = 0; g < 5; ++g) {
#pragma unroll
    for (int j = 0; j < 4; ++j) {
      int r = (g & 1) ? rotB[j] : rotA[j];
      x0 += x1;
      x1 = rotl32(x1, r) ^ x0;
    }
    x0 += ks[(g + 1) % 3];
    x1 += ks[(g + 2) % 3] + (unsigned)(g + 1);
  }
  unsigned bits = lowHalf ? x0 : x1;
  union { unsigned u; float f; } c;
  c.u = (bits >> 9) | 0x3f800000u;        // uniform in [1,2)
  return (c.f - 1.0f) < 0.5f;             // keep prob = 0.5
}

// ---------------- zero fill ----------------
__global__ void zero_f32(float* __restrict__ p, long n) {
  long i = (long)blockIdx.x * blockDim.x + threadIdx.x;
  long stride = (long)gridDim.x * blockDim.x;
  for (; i < n; i += stride) p[i] = 0.0f;
}

// ------- pre-swizzle stacked weights into WMMA B-fragment order (f16) -------
// B matrix for layer L: K=256 x Ncols; B[k][n] = k<128 ? Wl[n][k] : Wr[n][k-128]
// Fragment layout (16x16x32 f16 B): lane L holds col n=L&15, K-range base (L>>4)*16,
// element i -> K = kc*32 + (L>>4)*16 + i.  Stored as [ct][kc][lane][16] halves.
__global__ void prep_weights(const float* __restrict__ W1l, const float* __restrict__ W1r,
                             const float* __restrict__ W2l, const float* __restrict__ W2r,
                             unsigned short* __restrict__ B1, unsigned short* __restrict__ B2) {
  _Float16* b1h = reinterpret_cast<_Float16*>(B1);
  _Float16* b2h = reinterpret_cast<_Float16*>(B2);
  const int n1 = 8 * 8 * 32 * 16;   // 8 col-tiles (128 cols) x 8 k-chunks
  const int n2 = 4 * 8 * 32 * 16;   // 4 col-tiles (64 cols)  x 8 k-chunks
  int t = blockIdx.x * blockDim.x + threadIdx.x;
  int stride = gridDim.x * blockDim.x;
  for (; t < n1 + n2; t += stride) {
    if (t < n1) {
      int i = t & 15, lane = (t >> 4) & 31, kc = (t >> 9) & 7, ct = t >> 12;
      int n  = ct * 16 + (lane & 15);
      int kk = kc * 32 + (lane >> 4) * 16 + i;
      float v = (kk < DIN) ? W1l[n * DIN + kk] : W1r[n * DIN + (kk - DIN)];
      b1h[t] = (_Float16)v;
    } else {
      int t2 = t - n1;
      int i = t2 & 15, lane = (t2 >> 4) & 31, kc = (t2 >> 9) & 7, ct = t2 >> 12;
      int n  = ct * 16 + (lane & 15);
      int kk = kc * 32 + (lane >> 4) * 16 + i;
      float v = (kk < DH) ? W2l[n * DH + kk] : W2r[n * DH + (kk - DH)];
      b2h[t2] = (_Float16)v;
    }
  }
}

// ---------------- edge scatter: agg[dst] += feat[src], cnt[dst] += 1 ----------------
// One wave per edge; lane handles 4 consecutive floats (128 = 32*4).
// agg/feat/cnt all fit in the 192MB L2, so atomics stay on-chip.
__global__ void scatter_accum(const float* __restrict__ feat,
                              const long long* __restrict__ ei,
                              float* __restrict__ agg,
                              float* __restrict__ cnt,
                              int addCnt) {
  int wid  = blockIdx.x * (blockDim.x >> 5) + (threadIdx.x >> 5);
  int lane = threadIdx.x & 31;
  int nw   = gridDim.x * (blockDim.x >> 5);
  for (int e = wid; e < EE; e += nw) {
    int s = (int)ei[e];
    int d = (int)ei[EE + e];
    const float4 v = *reinterpret_cast<const float4*>(feat + (size_t)s * DH + lane * 4);
    float* dp = agg + (size_t)d * DH + lane * 4;
    __hip_atomic_fetch_add(dp + 0, v.x, __ATOMIC_RELAXED, __HIP_MEMORY_SCOPE_AGENT);
    __hip_atomic_fetch_add(dp + 1, v.y, __ATOMIC_RELAXED, __HIP_MEMORY_SCOPE_AGENT);
    __hip_atomic_fetch_add(dp + 2, v.z, __ATOMIC_RELAXED, __HIP_MEMORY_SCOPE_AGENT);
    __hip_atomic_fetch_add(dp + 3, v.w, __ATOMIC_RELAXED, __HIP_MEMORY_SCOPE_AGENT);
    if (addCnt && lane == 0)
      __hip_atomic_fetch_add(cnt + d, 1.0f, __ATOMIC_RELAXED, __HIP_MEMORY_SCOPE_AGENT);
  }
}

// ---------------- fused SAGE GEMM via v_wmma_f32_16x16x32_f16 ----------------
// Per block: 16 rows.  A = [mean | feat] (16 x 256) staged f16 in LDS.
// Wave w computes output cols [16w, 16w+16); 8 k-chunks of 32 -> 8 WMMAs.
// mode==1: +bias, relu, dropout(0.5), write h.  mode==0: +bias, write out.
__global__ void sage_gemm(const float* __restrict__ agg,
                          const float* __restrict__ cnt,
                          const float* __restrict__ feat,
                          const unsigned short* __restrict__ Bfrag,
                          const float* __restrict__ bias,
                          float* __restrict__ out,
                          int outDim, int mode) {
  __shared__ _Float16 As[16 * APITCH];
  __shared__ float rcp_s[16];
  const int tid = threadIdx.x;
  const int r0  = blockIdx.x * 16;

  if (tid < 16) {
    float c = cnt[r0 + tid];
    rcp_s[tid] = 1.0f / fmaxf(c, 1.0f);   // isolated nodes -> mean 0 (agg is 0)
  }
  __syncthreads();

  for (int idx = tid; idx < 16 * KTOT; idx += blockDim.x) {
    int m = idx >> 8, k = idx & 255;
    size_t row = (size_t)(r0 + m);
    float v = (k < DH) ? agg[row * DH + k] * rcp_s[m]
                       : feat[row * DH + (k - DH)];
    As[m * APITCH + k] = (_Float16)v;
  }
  __syncthreads();

  const int w = tid >> 5, lane = tid & 31;
  const int nloc = lane & 15, grp = lane >> 4;
  const _Float16* Bh = reinterpret_cast<const _Float16*>(Bfrag);

  v8f acc = {};
#pragma unroll
  for (int kc = 0; kc < 8; ++kc) {
    // A fragment: lane holds row m=lane&15; K contiguous to match B:
    // element i -> K = kc*32 + grp*16 + i
    v16h a;
    int kb = kc * 32 + grp * 16;
#pragma unroll
    for (int i = 0; i < 16; ++i) {
      int k = kb + i;
      a[i] = As[nloc * APITCH + k];
    }
    // B fragment: pre-swizzled, one aligned 32B load per lane
    v16h b = *reinterpret_cast<const v16h*>(Bh + ((size_t)(w * 8 + kc) * 32 + lane) * 16);
    acc = __builtin_amdgcn_wmma_f32_16x16x32_f16(false, a, false, b,
                                                 (short)0, acc, false, false);
  }

  int col = w * 16 + nloc;
  float bv = bias[col];
#pragma unroll
  for (int v = 0; v < 8; ++v) {
    int m = grp * 8 + v;
    size_t row = (size_t)(r0 + m);
    float val = acc[v] + bv;
    if (mode == 1) {
      val = fmaxf(val, 0.0f);
      unsigned idx = (unsigned)(row * DH + col);
      val = dropout_keep(idx) ? val * 2.0f : 0.0f;   // /(1-p), p=0.5
    }
    out[row * outDim + col] = val;
  }
}

// ---------------- launcher ----------------
extern "C" void kernel_launch(void* const* d_in, const int* in_sizes, int n_in,
                              void* d_out, int out_size, void* d_ws, size_t ws_size,
                              hipStream_t stream) {
  const float*     x   = (const float*)d_in[0];
  const long long* ei  = (const long long*)d_in[1];   // int64 edge_index [2,E]
  const float*     W1l = (const float*)d_in[2];
  const float*     b1  = (const float*)d_in[3];
  const float*     W1r = (const float*)d_in[4];
  const float*     W2l = (const float*)d_in[5];
  const float*     b2  = (const float*)d_in[6];
  const float*     W2r = (const float*)d_in[7];
  float* outp = (float*)d_out;

  char* ws = (char*)d_ws;
  const size_t AGG_B = (size_t)NN * DH * sizeof(float);   // 51.2 MB
  const size_t CNT_B = (size_t)NN * sizeof(float);        // 0.4 MB
  const size_t H_B   = (size_t)NN * DH * sizeof(float);   // 51.2 MB
  float* agg = (float*)ws;
  float* cnt = (float*)(ws + AGG_B);
  float* h   = (float*)(ws + AGG_B + CNT_B);
  unsigned short* B1 = (unsigned short*)(ws + AGG_B + CNT_B + H_B);  // 64 KB
  unsigned short* B2 = B1 + 8 * 8 * 32 * 16;                         // 32 KB

  // 1) zero agg + cnt (contiguous)
  zero_f32<<<2048, 256, 0, stream>>>(agg, (long)NN * DH + NN);
  // 2) pre-swizzle weights into WMMA fragment order
  prep_weights<<<256, 256, 0, stream>>>(W1l, W1r, W2l, W2r, B1, B2);
  // 3) layer-1 mean aggregation (with degree counts)
  scatter_accum<<<8192, 256, 0, stream>>>(x, ei, agg, cnt, 1);
  // 4) h = dropout(relu([mean|x] @ [W1l|W1r]^T + b1))
  sage_gemm<<<NN / 16, 256, 0, stream>>>(agg, cnt, x, B1, b1, h, DH, 1);
  // 5) re-zero agg (degrees unchanged, keep cnt)
  zero_f32<<<2048, 256, 0, stream>>>(agg, (long)NN * DH);
  // 6) layer-2 aggregation over h
  scatter_accum<<<8192, 256, 0, stream>>>(h, ei, agg, cnt, 0);
  // 7) out = [mean2|h] @ [W2l|W2r]^T + b2
  sage_gemm<<<NN / 16, 128, 0, stream>>>(agg, cnt, h, B2, b2, outp, DOUT, 0);
}
